// BitNetModel_18554258719008
// MI455X (gfx1250) — 8-bit matrix operand order repaired and checked by reading, compiled, not run on hardware
//
#include <hip/hip_runtime.h>
#include <stdint.h>

#define TOKENS 8192
#define DMODEL 2048
#define DFF    8192

#define LDS_STRIDE 80
#define TILE_BYTES (128 * LDS_STRIDE)
#define BUF_BYTES  (2 * TILE_BYTES)
#define NBUF 3

typedef int v8i __attribute__((ext_vector_type(8)));
typedef int v4i __attribute__((ext_vector_type(4)));
typedef int v2i __attribute__((ext_vector_type(2)));

__device__ __forceinline__ v8i load_a_frag(const int8_t* p) {
  v2i c0 = *(const v2i*)(p + 0);
  v2i c1 = *(const v2i*)(p + 8);
  v2i c2 = *(const v2i*)(p + 32);
  v2i c3 = *(const v2i*)(p + 40);
  v8i r;
  r[0] = c0[0]; r[1] = c0[1]; r[2] = c1[0]; r[3] = c1[1];
  r[4] = c2[0]; r[5] = c2[1]; r[6] = c3[0]; r[7] = c3[1];
  return r;
}

__device__ __forceinline__ v8i load_b_frag(const int8_t* p) {
  v4i q0 = *(const v4i*)(p + 0);
  v4i q1 = *(const v4i*)(p + 32);
  v8i r;
  r[0] = q0[0]; r[1] = q0[1]; r[2] = q0[2]; r[3] = q0[3];
  r[4] = q1[0]; r[5] = q1[1]; r[6] = q1[2]; r[7] = q1[3];
  return r;
}

__device__ __forceinline__ v8i wmma_iu8(v8i a, v8i b, v8i c) {
  return __builtin_amdgcn_wmma_i32_16x16x64_iu8(true, a, true, b, c, false, false);
}

#define ASYNC_LD16(ldsAddr, gPtr)                                              \
  asm volatile("global_load_async_to_lds_b128 %0, %1, off"                     \
               :: "v"(ldsAddr), "v"(gPtr) : "memory")
#define WAIT_ASYNC0() asm volatile("s_wait_asynccnt 0x0" ::: "memory")
#define WAIT_ASYNC4() asm volatile("s_wait_asynccnt 0x4" ::: "memory")

__global__ void __launch_bounds__(256) zero_scalars(unsigned* sc) {
  if (threadIdx.x < 8) sc[threadIdx.x] = 0u;
}

__global__ void __launch_bounds__(256) absmax_kernel(const float* __restrict__ x,
                                                     size_t n, unsigned* __restrict__ out) {
  size_t i = (size_t)blockIdx.x * 256 + threadIdx.x;
  size_t stride = (size_t)gridDim.x * 256;
  float m = 0.0f;
  for (; i < n; i += stride) m = fmaxf(m, fabsf(x[i]));
  __shared__ float s[256];
  s[threadIdx.x] = m;
  __syncthreads();
  for (int t = 128; t > 0; t >>= 1) {
    if (threadIdx.x < t) s[threadIdx.x] = fmaxf(s[threadIdx.x], s[threadIdx.x + t]);
    __syncthreads();
  }
  if (threadIdx.x == 0) atomicMax(out, __float_as_uint(s[0]));
}

__global__ void __launch_bounds__(256) abssum_partial(const float* __restrict__ w,
                                                      size_t n, float* __restrict__ partial) {
  size_t i = (size_t)blockIdx.x * 256 + threadIdx.x;
  size_t stride = (size_t)gridDim.x * 256;
  float a = 0.0f;
  for (; i < n; i += stride) a += fabsf(w[i]);
  __shared__ float s[256];
  s[threadIdx.x] = a;
  __syncthreads();
  for (int t = 128; t > 0; t >>= 1) {
    if (threadIdx.x < t) s[threadIdx.x] += s[threadIdx.x + t];
    __syncthreads();
  }
  if (threadIdx.x == 0) partial[blockIdx.x] = s[0];
}

__global__ void __launch_bounds__(256) finalize_means(const float* __restrict__ P1,
                                                      const float* __restrict__ P2,
                                                      float* __restrict__ scf,
                                                      int nb, float invn1, float invn2) {
  __shared__ float s[256];
  float a = 0.0f;
  for (int i = threadIdx.x; i < nb; i += 256) a += P1[i];
  s[threadIdx.x] = a;
  __syncthreads();
  for (int t = 128; t > 0; t >>= 1) {
    if (threadIdx.x < t) s[threadIdx.x] += s[threadIdx.x + t];
    __syncthreads();
  }
  if (threadIdx.x == 0) scf[1] = s[0] * invn1;
  __syncthreads();
  float b = 0.0f;
  for (int i = threadIdx.x; i < nb; i += 256) b += P2[i];
  s[threadIdx.x] = b;
  __syncthreads();
  for (int t = 128; t > 0; t >>= 1) {
    if (threadIdx.x < t) s[threadIdx.x] += s[threadIdx.x + t];
    __syncthreads();
  }
  if (threadIdx.x == 0) scf[2] = s[0] * invn2;
}

__global__ void __launch_bounds__(256) quant_act(const float* __restrict__ x,
                                                 int8_t* __restrict__ q,
                                                 const unsigned* __restrict__ sc,
                                                 int idx, size_t n4) {
  size_t i = (size_t)blockIdx.x * 256 + threadIdx.x;
  if (i >= n4) return;
  float s = 127.0f / __uint_as_float(sc[idx]);
  float4 v = ((const float4*)x)[i];
  char4 o;
  o.x = (signed char)__float2int_rn(fminf(fmaxf(v.x * s, -127.0f), 127.0f));
  o.y = (signed char)__float2int_rn(fminf(fmaxf(v.y * s, -127.0f), 127.0f));
  o.z = (signed char)__float2int_rn(fminf(fmaxf(v.z * s, -127.0f), 127.0f));
  o.w = (signed char)__float2int_rn(fminf(fmaxf(v.w * s, -127.0f), 127.0f));
  ((char4*)q)[i] = o;
}

__global__ void __launch_bounds__(256) quant_wt(const float* __restrict__ w,
                                                int8_t* __restrict__ q,
                                                const float* __restrict__ scf,
                                                int idx, size_t n4) {
  size_t i = (size_t)blockIdx.x * 256 + threadIdx.x;
  if (i >= n4) return;
  float s = 1.0f / scf[idx];
  float4 v = ((const float4*)w)[i];
  char4 o;
  o.x = (signed char)__float2int_rn(fminf(fmaxf(v.x * s, -1.0f), 1.0f));
  o.y = (signed char)__float2int_rn(fminf(fmaxf(v.y * s, -1.0f), 1.0f));
  o.z = (signed char)__float2int_rn(fminf(fmaxf(v.z * s, -1.0f), 1.0f));
  o.w = (signed char)__float2int_rn(fminf(fmaxf(v.w * s, -1.0f), 1.0f));
  ((char4*)q)[i] = o;
}

template <bool GELU_OUT>
__global__ void __launch_bounds__(256)
gemm_ternary_iu8(const int8_t* __restrict__ A, const int8_t* __restrict__ B,
                 const float* __restrict__ bias, float* __restrict__ Out,
                 const unsigned* __restrict__ sc, int maxIdx, int meanIdx,
                 unsigned* __restrict__ maxOut, int N, int K) {
  __shared__ int8_t lds[NBUF * BUF_BYTES];

  const int t    = threadIdx.x;
  const int lane = t & 31;
  const int wave = t >> 5;
  const int l15  = lane & 15;
  const int lh   = lane >> 4;
  const int wm   = wave >> 1;
  const int wn   = wave & 1;

  const int nBlkN = N >> 7;
  const int M0 = (blockIdx.x / nBlkN) * 128;
  const int N0 = (blockIdx.x % nBlkN) * 128;

  const int u0 = t, u1 = t + 256;
  const int r0 = u0 >> 2, c0 = (u0 & 3) * 16;
  const int r1 = u1 >> 2, c1 = (u1 & 3) * 16;
  const int8_t* gA0 = A + (size_t)(M0 + r0) * K + c0;
  const int8_t* gA1 = A + (size_t)(M0 + r1) * K + c1;
  const int8_t* gB0 = B + (size_t)(N0 + r0) * K + c0;
  const int8_t* gB1 = B + (size_t)(N0 + r1) * K + c1;

  const unsigned ldsBase = (unsigned)(size_t)(&lds[0]);
  const unsigned wA0 = ldsBase + (unsigned)(r0 * LDS_STRIDE + c0);
  const unsigned wA1 = ldsBase + (unsigned)(r1 * LDS_STRIDE + c1);
  const unsigned wB0 = ldsBase + TILE_BYTES + (unsigned)(r0 * LDS_STRIDE + c0);
  const unsigned wB1 = ldsBase + TILE_BYTES + (unsigned)(r1 * LDS_STRIDE + c1);

  const int paOff = (wm * 32 + l15) * LDS_STRIDE + lh * 16;
  const int pbOff = TILE_BYTES + (wn * 64 + l15) * LDS_STRIDE + lh * 16;

  v8i acc00 = {}, acc01 = {}, acc02 = {}, acc03 = {};
  v8i acc10 = {}, acc11 = {}, acc12 = {}, acc13 = {};

  ASYNC_LD16(wA0, gA0);
  ASYNC_LD16(wA1, gA1);
  ASYNC_LD16(wB0, gB0);
  ASYNC_LD16(wB1, gB1);
  gA0 += 64; gA1 += 64; gB0 += 64; gB1 += 64;
  ASYNC_LD16(wA0 + BUF_BYTES, gA0);
  ASYNC_LD16(wA1 + BUF_BYTES, gA1);
  ASYNC_LD16(wB0 + BUF_BYTES, gB0);
  ASYNC_LD16(wB1 + BUF_BYTES, gB1);
  WAIT_ASYNC4();
  __syncthreads();

  unsigned rdOff = 0;
  unsigned wrOff = 2 * BUF_BYTES;
  for (int k0 = 0; k0 < K; k0 += 64) {
    const bool issue2 = (k0 + 128) < K;
    if (issue2) {
      gA0 += 64; gA1 += 64; gB0 += 64; gB1 += 64;
      ASYNC_LD16(wA0 + wrOff, gA0);
      ASYNC_LD16(wA1 + wrOff, gA1);
      ASYNC_LD16(wB0 + wrOff, gB0);
      ASYNC_LD16(wB1 + wrOff, gB1);
    }

    const int8_t* pa = lds + rdOff + paOff;
    const int8_t* pb = lds + rdOff + pbOff;
    v8i a0 = load_a_frag(pa);
    v8i a1 = load_a_frag(pa + 16 * LDS_STRIDE);
    v8i b0 = load_b_frag(pb);
    v8i b1 = load_b_frag(pb + 16 * LDS_STRIDE);
    acc00 = wmma_iu8(a0, b0, acc00);
    acc10 = wmma_iu8(a1, b0, acc10);
    v8i b2 = load_b_frag(pb + 32 * LDS_STRIDE);
    acc01 = wmma_iu8(a0, b1, acc01);
    acc11 = wmma_iu8(a1, b1, acc11);
    v8i b3 = load_b_frag(pb + 48 * LDS_STRIDE);
    acc02 = wmma_iu8(a0, b2, acc02);
    acc12 = wmma_iu8(a1, b2, acc12);
    acc03 = wmma_iu8(a0, b3, acc03);
    acc13 = wmma_iu8(a1, b3, acc13);

    if (issue2) WAIT_ASYNC4();
    else        WAIT_ASYNC0();
    __syncthreads();
    rdOff = (rdOff == (NBUF - 1) * BUF_BYTES) ? 0u : rdOff + BUF_BYTES;
    wrOff = (wrOff == (NBUF - 1) * BUF_BYTES) ? 0u : wrOff + BUF_BYTES;
  }

  const float* scf = (const float*)sc;
  const float alpha = scf[meanIdx] * (__uint_as_float(sc[maxIdx]) * (1.0f / 127.0f));
  float lmax = 0.0f;

#define EPI(ACC, MT, NT)                                                       \
  {                                                                            \
    const int n = N0 + wn * 64 + (NT)*16 + l15;                                \
    const float bv = bias[n];                                                  \
    _Pragma("unroll") for (int r = 0; r < 8; ++r) {                            \
      const int m = M0 + wm * 32 + (MT)*16 + lh * 8 + r;                       \
      float v = (float)(ACC)[r] * alpha + bv;                                  \
      if (GELU_OUT) {                                                          \
        v = 0.5f * v * (1.0f + erff(v * 0.70710678118654752f));                \
        lmax = fmaxf(lmax, fabsf(v));                                          \
      }                                                                        \
      Out[(size_t)m * N + n] = v;                                              \
    }                                                                          \
  }

  EPI(acc00, 0, 0) EPI(acc01, 0, 1) EPI(acc02, 0, 2) EPI(acc03, 0, 3)
  EPI(acc10, 1, 0) EPI(acc11, 1, 1) EPI(acc12, 1, 2) EPI(acc13, 1, 3)
#undef EPI

  if (GELU_OUT) {
    float* sred = (float*)lds;
    sred[t] = lmax;
    __syncthreads();
    for (int st = 128; st > 0; st >>= 1) {
      if (t < st) sred[t] = fmaxf(sred[t], sred[t + st]);
      __syncthreads();
    }
    if (t == 0) atomicMax(maxOut, __float_as_uint(sred[0]));
  }
}

extern "C" void kernel_launch(void* const* d_in, const int* in_sizes, int n_in,
                              void* d_out, int out_size, void* d_ws, size_t ws_size,
                              hipStream_t stream) {
  const float* x  = (const float*)d_in[0];
  const float* W1 = (const float*)d_in[1];
  const float* b1 = (const float*)d_in[2];
  const float* W2 = (const float*)d_in[3];
  const float* b2 = (const float*)d_in[4];
  float* out = (float*)d_out;

  char* ws = (char*)d_ws;
  unsigned* sc  = (unsigned*)(ws + 0);
  float*    scf = (float*)sc;
  float*    P1  = (float*)(ws + 1024);
  float*    P2  = (float*)(ws + 1024 + 4096);
  const size_t SZ_XW = (size_t)DFF * DMODEL;
  int8_t* XQ  = (int8_t*)(ws + 16384);
  int8_t* W1Q = XQ  + SZ_XW;
  int8_t* W2Q = W1Q + SZ_XW;
  int8_t* HQ  = W2Q + SZ_XW;
  float*  H   = (float*)(HQ + (size_t)TOKENS * DFF);

  zero_scalars<<<1, 256, 0, stream>>>(sc);
  absmax_kernel<<<1024, 256, 0, stream>>>(x, (size_t)TOKENS * DMODEL, &sc[0]);
  abssum_partial<<<1024, 256, 0, stream>>>(W1, SZ_XW, P1);
  abssum_partial<<<1024, 256, 0, stream>>>(W2, SZ_XW, P2);
  finalize_means<<<1, 256, 0, stream>>>(P1, P2, scf, 1024,
                                        1.0f / (float)SZ_XW, 1.0f / (float)SZ_XW);

  quant_act<<<(TOKENS * DMODEL / 4) / 256, 256, 0, stream>>>(x, XQ, sc, 0,
                                                             (size_t)TOKENS * DMODEL / 4);
  quant_wt<<<(int)(SZ_XW / 4 / 256), 256, 0, stream>>>(W1, W1Q, scf, 1, SZ_XW / 4);
  quant_wt<<<(int)(SZ_XW / 4 / 256), 256, 0, stream>>>(W2, W2Q, scf, 2, SZ_XW / 4);

  gemm_ternary_iu8<true><<<4096, 256, 0, stream>>>(XQ, W1Q, b1, H, sc, 0, 1,
                                                   &sc[3], DFF, DMODEL);

  quant_act<<<(int)((size_t)TOKENS * DFF / 4 / 256), 256, 0, stream>>>(
      H, HQ, sc, 3, (size_t)TOKENS * DFF / 4);

  gemm_ternary_iu8<false><<<1024, 256, 0, stream>>>(HQ, W2Q, b2, out, sc, 3, 2,
                                                    nullptr, DMODEL, DFF);
}
